// StructureModule_80092550136290
// MI455X (gfx1250) — matrix operand order repaired and checked by reading, not compiled, not run on hardware
//
#include <hip/hip_runtime.h>
#include <stdint.h>

typedef float  v2f   __attribute__((ext_vector_type(2)));
typedef float  v8f   __attribute__((ext_vector_type(8)));
typedef __bf16 v2bf  __attribute__((ext_vector_type(2)));
typedef __bf16 v16bf __attribute__((ext_vector_type(16)));
typedef __attribute__((address_space(3))) __bf16 lds_bf16_t;

#define NRES 512
#define HH 12
#define POS_SCALE 10.0f
#define EPS_LN 1e-6f
#define DIST_EPS 1e-8f
#define SCALAR_W 0.14433756729740643f   /* (1/48)^0.5  */
#define POINT_W0 0.13608276348795434f   /* (1/54)^0.5  */
#define PAIR_W   0.57735026918962576f   /* (1/3)^0.5   */

#define CDIV(a, b) (((a) + (b) - 1) / (b))

// ---------------------------------------------------------------------------
// Generic f32 GEMM: C = act( A(MxK) @ B(KxN | NxK if transB) + bias ).
// One wave = one 16x16 C tile via v_wmma_f32_16x16x4_f32; 4 waves/block.
// Requires K % 4 == 0 (all call sites).  Out-of-range rows/cols are handled
// by CLAMPING the loop-invariant index: garbage A-rows / B-cols only affect
// C rows/cols that are never stored -> no per-k guards, no exec divergence.
// ---------------------------------------------------------------------------
__global__ void k_gemm_f32(const float* __restrict__ A, const float* __restrict__ B,
                           const float* __restrict__ bias, float* __restrict__ C,
                           int M, int N, int K, int lda, int ldb, int ldc,
                           int64_t sA, int64_t sB, int64_t sC, int transB, int relu)
{
    int lane = threadIdx.x & 31;
    int wv   = threadIdx.x >> 5;
    int half = lane >> 4;
    int lr   = lane & 15;
    int row0 = blockIdx.x * 16;
    int col0 = blockIdx.y * 64 + wv * 16;
    if (row0 >= M || col0 >= N) return;
    int64_t z = blockIdx.z;
    A += z * sA; B += z * sB; C += z * sC;

    int mc = row0 + lr; if (mc > M - 1) mc = M - 1;   // clamp, see header comment
    int nc = col0 + lr; if (nc > N - 1) nc = N - 1;

    v8f acc = {0.f, 0.f, 0.f, 0.f, 0.f, 0.f, 0.f, 0.f};
    const float* pA = A + (int64_t)mc * lda + half * 2;

    if (transB) {
        const float* pB = B + (int64_t)nc * ldb + half * 2;
#pragma unroll 4
        for (int k0 = 0; k0 < K; k0 += 4) {
            v2f a = *(const v2f*)pA;
            v2f b = *(const v2f*)pB;
            acc = __builtin_amdgcn_wmma_f32_16x16x4_f32(false, a, false, b,
                                                        (short)0, acc, false, false);
            pA += 4; pB += 4;
        }
    } else {
        const float* pB = B + (int64_t)(half * 2) * ldb + nc;
        int64_t bstep = (int64_t)4 * ldb;
#pragma unroll 4
        for (int k0 = 0; k0 < K; k0 += 4) {
            v2f a = *(const v2f*)pA;
            v2f b; b[0] = pB[0]; b[1] = pB[ldb];
            acc = __builtin_amdgcn_wmma_f32_16x16x4_f32(false, a, false, b,
                                                        (short)0, acc, false, false);
            pA += 4; pB += bstep;
        }
    }
#pragma unroll
    for (int r = 0; r < 8; ++r) {
        int mm = row0 + half * 8 + r;
        int nn = col0 + lr;
        if (mm < M && nn < N) {
            float v = acc[r] + (bias ? bias[nn] : 0.f);
            if (relu) v = fmaxf(v, 0.f);
            C[(int64_t)mm * ldc + nn] = v;
        }
    }
}

// ---------------------------------------------------------------------------
// Generic bf16 GEMM (f32 acc): v_wmma_f32_16x16x32_bf16, K % 32 == 0.
// Same clamping strategy. Used for the layer-invariant pair bias (B = tiny
// 128x12 weight, L0-resident, so strided column reads are fine).
// ---------------------------------------------------------------------------
__global__ void k_gemm_bf16(const __bf16* __restrict__ A, const __bf16* __restrict__ B,
                            const float* __restrict__ bias, float* __restrict__ C,
                            int M, int N, int K, int lda, int ldb, int ldc,
                            int64_t sA, int64_t sB, int64_t sC)
{
    int lane = threadIdx.x & 31;
    int wv   = threadIdx.x >> 5;
    int half = lane >> 4;
    int lr   = lane & 15;
    int row0 = blockIdx.x * 16;
    int col0 = blockIdx.y * 64 + wv * 16;
    if (row0 >= M || col0 >= N) return;
    int64_t z = blockIdx.z;
    A += z * sA; B += z * sB; C += z * sC;

    int mc = row0 + lr; if (mc > M - 1) mc = M - 1;
    int nc = col0 + lr; if (nc > N - 1) nc = N - 1;

    v8f acc = {0.f, 0.f, 0.f, 0.f, 0.f, 0.f, 0.f, 0.f};
    const __bf16* pA = A + (int64_t)mc * lda + half * 8;

    for (int k0 = 0; k0 < K; k0 += 32) {
        v16bf a, b;
#pragma unroll
        for (int v = 0; v < 8; ++v) {
            v2bf t = *(const v2bf*)(pA + k0 + (v >> 2) * 16 + (v & 3) * 2);
            a[2 * v]     = t.x;
            a[2 * v + 1] = t.y;
        }
        const __bf16* pb = B + (int64_t)(k0 + half * 8) * ldb + nc;
#pragma unroll
        for (int e = 0; e < 16; ++e) b[e] = pb[(int64_t)((e >> 3) * 16 + (e & 7)) * ldb];
        acc = __builtin_amdgcn_wmma_f32_16x16x32_bf16(false, a, false, b,
                                                      (short)0, acc, false, false);
    }
#pragma unroll
    for (int r = 0; r < 8; ++r) {
        int mm = row0 + half * 8 + r;
        int nn = col0 + lr;
        if (mm < M && nn < N) {
            float v = acc[r] + (bias ? bias[nn] : 0.f);
            C[(int64_t)mm * ldc + nn] = v;
        }
    }
}

// ---------------------------------------------------------------------------
// Specialized res_2d = attn[n](12x512) @ pair_n[n](512x128), bf16 WMMA.
// The pair K-tile (32x16 bf16 per wave) is staged into per-wave LDS with
// CDNA5 async loads (global_load_async_to_lds_b128), synchronized with
// s_wait_asynccnt. grid = (2, NRES), block = 128 (4 waves x 16 cols).
// ---------------------------------------------------------------------------
__global__ void k_res2d_async(const __bf16* __restrict__ attnb,
                              const __bf16* __restrict__ pairb,
                              float* __restrict__ res2d)
{
    __shared__ __bf16 tile[4][32][16];   // [wave][k][col]
    int nb   = blockIdx.y;
    int lane = threadIdx.x & 31;
    int wv   = threadIdx.x >> 5;
    int half = lane >> 4;
    int lr   = lane & 15;
    int col0 = blockIdx.x * 64 + wv * 16;

    // LDS byte offset of this lane's staging row (addrspace(3) => LDS offset)
    uint32_t ldsRow = (uint32_t)(uintptr_t)(lds_bf16_t*)&tile[wv][lane][0];

    int ha = lr < HH ? lr : HH - 1;      // clamp A row (rows 12..15 unstored)
    const __bf16* pA = attnb + (int64_t)nb * HH * NRES + (int64_t)ha * NRES + half * 8;
    const __bf16* pB = pairb + (int64_t)nb * NRES * 128 + col0;

    v8f acc = {0.f, 0.f, 0.f, 0.f, 0.f, 0.f, 0.f, 0.f};
    for (int k0 = 0; k0 < NRES; k0 += 32) {
        // stage: lane copies row (k0+lane), 16 bf16 = 32B, as 2 async b128 ops
        const __bf16* g = pB + (int64_t)(k0 + lane) * 128;
        asm volatile("global_load_async_to_lds_b128 %0, %1, off"
                     :: "v"(ldsRow), "v"(g) : "memory");
        asm volatile("global_load_async_to_lds_b128 %0, %1, off offset:16"
                     :: "v"(ldsRow), "v"(g) : "memory");
        asm volatile("s_wait_asynccnt 0x0" ::: "memory");

        v16bf a, b;
#pragma unroll
        for (int v = 0; v < 8; ++v) {
            v2bf t = *(const v2bf*)(pA + k0 + (v >> 2) * 16 + (v & 3) * 2);
            a[2 * v]     = t.x;
            a[2 * v + 1] = t.y;
        }
#pragma unroll
        for (int e = 0; e < 16; ++e) b[e] = tile[wv][half * 8 + (e >> 3) * 16 + (e & 7)][lr];
        acc = __builtin_amdgcn_wmma_f32_16x16x32_bf16(false, a, false, b,
                                                      (short)0, acc, false, false);
    }
    float* Cb = res2d + (int64_t)nb * 1536;
#pragma unroll
    for (int r = 0; r < 8; ++r) {
        int mm = half * 8 + r;
        if (mm < HH) Cb[mm * 128 + col0 + lr] = acc[r];
    }
}

// ---------------------------------------------------------------------------
// LayerNorm (optionally y = LN(x + d)). blockDim = 128, one row per block.
// ---------------------------------------------------------------------------
__global__ void k_ln(const float* __restrict__ x, const float* __restrict__ d,
                     const float* __restrict__ s, const float* __restrict__ b,
                     float* __restrict__ y, int D)
{
    int row = blockIdx.x;
    const float* xr = x + (int64_t)row * D;
    const float* dr = d ? d + (int64_t)row * D : nullptr;
    __shared__ float shs[128], shq[128];
    float sum = 0.f, sq = 0.f;
    for (int i = threadIdx.x; i < D; i += 128) {
        float v = xr[i] + (dr ? dr[i] : 0.f);
        sum += v; sq += v * v;
    }
    shs[threadIdx.x] = sum; shq[threadIdx.x] = sq;
    __syncthreads();
    for (int o = 64; o > 0; o >>= 1) {
        if (threadIdx.x < (unsigned)o) {
            shs[threadIdx.x] += shs[threadIdx.x + o];
            shq[threadIdx.x] += shq[threadIdx.x + o];
        }
        __syncthreads();
    }
    float mean = shs[0] / D;
    float var  = shq[0] / D - mean * mean;
    float inv  = rsqrtf(var + EPS_LN);
    for (int i = threadIdx.x; i < D; i += 128) {
        float v = xr[i] + (dr ? dr[i] : 0.f);
        y[(int64_t)row * D + i] = (v - mean) * inv * s[i] + b[i];
    }
}

// pair LayerNorm (D=128) straight to bf16 (67MB bf16 pair stays L2-resident)
__global__ void k_ln_pair_bf16(const float* __restrict__ x, const float* __restrict__ s,
                               const float* __restrict__ b, __bf16* __restrict__ y)
{
    int row = blockIdx.x;
    int t = threadIdx.x; // 128
    const float* xr = x + (int64_t)row * 128;
    float v = xr[t];
    __shared__ float sh[128], sq[128];
    sh[t] = v; sq[t] = v * v;
    __syncthreads();
    for (int o = 64; o > 0; o >>= 1) {
        if (t < o) { sh[t] += sh[t + o]; sq[t] += sq[t + o]; }
        __syncthreads();
    }
    float mean = sh[0] * (1.f / 128.f);
    float var  = sq[0] * (1.f / 128.f) - mean * mean;
    float inv  = rsqrtf(var + EPS_LN);
    y[(int64_t)row * 128 + t] = (__bf16)((v - mean) * inv * s[t] + b[t]);
}

__global__ void k_cvt_bf16(const float* __restrict__ x, __bf16* __restrict__ y, int n)
{
    int i = blockIdx.x * blockDim.x + threadIdx.x;
    if (i < n) y[i] = (__bf16)x[i];
}

__global__ void k_add(float* __restrict__ y, const float* __restrict__ x, int n)
{
    int i = blockIdx.x * blockDim.x + threadIdx.x;
    if (i < n) y[i] += x[i];
}

__global__ void k_relu_copy(float* __restrict__ y, const float* __restrict__ x, int n)
{
    int i = blockIdx.x * blockDim.x + threadIdx.x;
    if (i < n) y[i] = fmaxf(x[i], 0.f);
}

// local points -> global frame, in place. pts: [n][P][3]
__global__ void k_pts_global(float* __restrict__ pts, const float* __restrict__ rot,
                             const float* __restrict__ trans, int P)
{
    int idx = blockIdx.x * blockDim.x + threadIdx.x;
    if (idx >= NRES * P) return;
    int n = idx / P, p = idx - n * P;
    float* v = pts + ((int64_t)n * P + p) * 3;
    float lx = v[0], ly = v[1], lz = v[2];
    const float* R = rot + n * 9;
    v[0] = R[0] * lx + R[1] * ly + R[2] * lz + trans[n * 3 + 0];
    v[1] = R[3] * lx + R[4] * ly + R[5] * lz + trans[n * 3 + 1];
    v[2] = R[6] * lx + R[7] * ly + R[8] * lz + trans[n * 3 + 2];
}

// y[i] = sum_{k<12} x[i*12+k]^2   (qq / kk point norms per (n,h))
__global__ void k_sqsum12(const float* __restrict__ x, float* __restrict__ y, int count)
{
    int i = blockIdx.x * blockDim.x + threadIdx.x;
    if (i >= count) return;
    float s = 0.f;
#pragma unroll
    for (int k = 0; k < 12; ++k) { float v = x[i * 12 + k]; s += v * v; }
    y[i] = s;
}

// fused logits finalize + softmax: 1 wave per (n,h); 16 keys per lane.
__global__ void k_softmax(const float* __restrict__ logS, const float* __restrict__ qkP,
                          const float* __restrict__ qq, const float* __restrict__ kk,
                          const float* __restrict__ b2d, const float* __restrict__ tpw,
                          const float* __restrict__ mask,
                          float* __restrict__ attn, __bf16* __restrict__ attnb)
{
    int nh = blockIdx.x;
    int n = nh / HH, h = nh - n * HH;
    int lane = threadIdx.x;
    float t  = tpw[h];
    float sp = (t > 20.f) ? t : log1pf(expf(t));
    float pw = POINT_W0 * sp;
    float qv = qq[nh];
    float mn = mask[n];
    const float* ls = logS + (int64_t)h * NRES * NRES + (int64_t)n * NRES;
    const float* qk = qkP  + (int64_t)h * NRES * NRES + (int64_t)n * NRES;
    const float* bd = b2d  + (int64_t)n * NRES * HH;
    float v[16];
    float mx = -3.0e38f;
#pragma unroll
    for (int i = 0; i < 16; ++i) {
        int m = i * 32 + lane;
        float d2  = qv + kk[m * HH + h] - 2.f * qk[m];
        float val = SCALAR_W * ls[m] - 0.5f * pw * d2
                  + PAIR_W * bd[m * HH + h]
                  + 1.0e9f * (mn * mask[m] - 1.f);
        v[i] = val;
        mx = fmaxf(mx, val);
    }
    for (int o = 16; o > 0; o >>= 1) mx = fmaxf(mx, __shfl_xor(mx, o, 32));
    float sum = 0.f;
#pragma unroll
    for (int i = 0; i < 16; ++i) { v[i] = expf(v[i] - mx); sum += v[i]; }
    for (int o = 16; o > 0; o >>= 1) sum += __shfl_xor(sum, o, 32);
    float inv = 1.f / sum;
    float*  ao = attn  + ((int64_t)n * HH + h) * NRES;
    __bf16* ab = attnb + ((int64_t)n * HH + h) * NRES;
#pragma unroll
    for (int i = 0; i < 16; ++i) {
        int m = i * 32 + lane;
        float a = v[i] * inv;
        ao[m] = a;
        ab[m] = (__bf16)a;
    }
}

// assemble feat[n,2112] = [res_s(192) | res_pl(288) | res_pn(96) | res_2d(1536)]
__global__ void k_feat(const float* __restrict__ resS, const float* __restrict__ resPg,
                       const float* __restrict__ res2d, const float* __restrict__ rot,
                       const float* __restrict__ trans, float* __restrict__ feat)
{
    int n = blockIdx.x;
    float* f = feat + (int64_t)n * 2112;
    for (int i = threadIdx.x; i < 192; i += 256) {
        int h = i >> 4, c = i & 15;
        f[i] = resS[(int64_t)h * NRES * 16 + n * 16 + c];
    }
    for (int i = threadIdx.x; i < 96; i += 256) {
        int h = i >> 3, p = i & 7;
        const float* g = resPg + (int64_t)h * NRES * 24 + n * 24 + p * 3;
        float dx = g[0] - trans[n * 3 + 0];
        float dy = g[1] - trans[n * 3 + 1];
        float dz = g[2] - trans[n * 3 + 2];
        const float* R = rot + n * 9;
        float lx = R[0] * dx + R[3] * dy + R[6] * dz;  // R^T
        float ly = R[1] * dx + R[4] * dy + R[7] * dz;
        float lz = R[2] * dx + R[5] * dy + R[8] * dz;
        f[192 + i * 3 + 0] = lx;
        f[192 + i * 3 + 1] = ly;
        f[192 + i * 3 + 2] = lz;
        f[480 + i] = sqrtf(lx * lx + ly * ly + lz * lz + DIST_EPS);
    }
    for (int i = threadIdx.x; i < 1536; i += 256)
        f[576 + i] = res2d[(int64_t)n * 1536 + i];
}

// small dense (N<16): y[n,j] = x[n,:] @ w[:,j] + b[j]
__global__ void k_small_dense(const float* __restrict__ x, const float* __restrict__ w,
                              const float* __restrict__ b, float* __restrict__ y,
                              int D, int Nout)
{
    int idx = blockIdx.x * blockDim.x + threadIdx.x;
    if (idx >= NRES * Nout) return;
    int n = idx / Nout, j = idx - n * Nout;
    float s = b[j];
    for (int i = 0; i < D; ++i) s += x[(int64_t)n * D + i] * w[(int64_t)i * Nout + j];
    y[idx] = s;
}

__global__ void k_init_pose(float* quat, float* rot, float* trans)
{
    int n = blockIdx.x * blockDim.x + threadIdx.x;
    if (n >= NRES) return;
    quat[n * 4 + 0] = 1.f; quat[n * 4 + 1] = 0.f; quat[n * 4 + 2] = 0.f; quat[n * 4 + 3] = 0.f;
    for (int k = 0; k < 9; ++k) rot[n * 9 + k] = (k % 4 == 0) ? 1.f : 0.f;
    trans[n * 3 + 0] = 0.f; trans[n * 3 + 1] = 0.f; trans[n * 3 + 2] = 0.f;
}

// quaternion pre_compose + traj write (traj translation scaled by POS_SCALE)
__global__ void k_compose(float* __restrict__ quat, float* __restrict__ rot,
                          float* __restrict__ trans, const float* __restrict__ upd,
                          float* __restrict__ traj)
{
    int n = blockIdx.x * blockDim.x + threadIdx.x;
    if (n >= NRES) return;
    float w = quat[n * 4 + 0], x = quat[n * 4 + 1], y = quat[n * 4 + 2], z = quat[n * 4 + 3];
    float bx = upd[n * 6 + 0], by = upd[n * 6 + 1], bz = upd[n * 6 + 2];
    float tx = upd[n * 6 + 3], ty = upd[n * 6 + 4], tz = upd[n * 6 + 5];
    float nw = w - (x * bx + y * by + z * bz);
    float nx = x + (w * bx + y * bz - z * by);
    float ny = y + (w * by + z * bx - x * bz);
    float nz = z + (w * bz + x * by - y * bx);
    float inv = rsqrtf(nw * nw + nx * nx + ny * ny + nz * nz + DIST_EPS);
    nw *= inv; nx *= inv; ny *= inv; nz *= inv;
    float* R = rot + n * 9;
    float t0 = trans[n * 3 + 0] + R[0] * tx + R[1] * ty + R[2] * tz;   // old rot
    float t1 = trans[n * 3 + 1] + R[3] * tx + R[4] * ty + R[5] * tz;
    float t2 = trans[n * 3 + 2] + R[6] * tx + R[7] * ty + R[8] * tz;
    R[0] = 1.f - 2.f * (ny * ny + nz * nz); R[1] = 2.f * (nx * ny - nw * nz); R[2] = 2.f * (nx * nz + nw * ny);
    R[3] = 2.f * (nx * ny + nw * nz); R[4] = 1.f - 2.f * (nx * nx + nz * nz); R[5] = 2.f * (ny * nz - nw * nx);
    R[6] = 2.f * (nx * nz - nw * ny); R[7] = 2.f * (ny * nz + nw * nx); R[8] = 1.f - 2.f * (nx * nx + ny * ny);
    quat[n * 4 + 0] = nw; quat[n * 4 + 1] = nx; quat[n * 4 + 2] = ny; quat[n * 4 + 3] = nz;
    trans[n * 3 + 0] = t0; trans[n * 3 + 1] = t1; trans[n * 3 + 2] = t2;
    traj[n * 7 + 0] = nw; traj[n * 7 + 1] = nx; traj[n * 7 + 2] = ny; traj[n * 7 + 3] = nz;
    traj[n * 7 + 4] = t0 * POS_SCALE; traj[n * 7 + 5] = t1 * POS_SCALE; traj[n * 7 + 6] = t2 * POS_SCALE;
}

// sidechain frames + atom14 positions (final layer only)
__global__ void k_sidechain(const float* __restrict__ un, const int* __restrict__ aatype,
                            const float* __restrict__ dframe, const int* __restrict__ a14g,
                            const float* __restrict__ a14p, const float* __restrict__ a14m,
                            const float* __restrict__ rot, const float* __restrict__ trans,
                            float* __restrict__ ang_out, float* __restrict__ pos_out)
{
    int n = blockIdx.x * blockDim.x + threadIdx.x;
    if (n >= NRES) return;
    int aa = aatype[n];
    float sg[8], cg[8];
    sg[0] = 0.f; cg[0] = 1.f;
#pragma unroll
    for (int g = 1; g < 8; ++g) {
        float ux = un[n * 14 + (g - 1) * 2 + 0];
        float uy = un[n * 14 + (g - 1) * 2 + 1];
        float inv = rsqrtf(ux * ux + uy * uy + EPS_LN);
        sg[g] = ux * inv; cg[g] = uy * inv;
        ang_out[n * 14 + (g - 1) * 2 + 0] = sg[g];
        ang_out[n * 14 + (g - 1) * 2 + 1] = cg[g];
    }
    float Rl[8][9], Tl[8][3];
    for (int g = 0; g < 8; ++g) {
        const float* M = dframe + (int64_t)(aa * 8 + g) * 16;
        for (int r = 0; r < 3; ++r) {
            Rl[g][r * 3 + 0] = M[r * 4 + 0];
            Rl[g][r * 3 + 1] =  cg[g] * M[r * 4 + 1] + sg[g] * M[r * 4 + 2];
            Rl[g][r * 3 + 2] = -sg[g] * M[r * 4 + 1] + cg[g] * M[r * 4 + 2];
            Tl[g][r] = M[r * 4 + 3];
        }
    }
    for (int g = 5; g < 8; ++g) {
        float R[9], T[3];
        for (int i = 0; i < 3; ++i) {
            for (int j = 0; j < 3; ++j)
                R[i * 3 + j] = Rl[g - 1][i * 3 + 0] * Rl[g][0 + j]
                             + Rl[g - 1][i * 3 + 1] * Rl[g][3 + j]
                             + Rl[g - 1][i * 3 + 2] * Rl[g][6 + j];
            T[i] = Rl[g - 1][i * 3 + 0] * Tl[g][0] + Rl[g - 1][i * 3 + 1] * Tl[g][1]
                 + Rl[g - 1][i * 3 + 2] * Tl[g][2] + Tl[g - 1][i];
        }
        for (int k = 0; k < 9; ++k) Rl[g][k] = R[k];
        for (int k = 0; k < 3; ++k) Tl[g][k] = T[k];
    }
    const float* Rn = rot + n * 9;
    float tw[3] = { trans[n * 3 + 0] * POS_SCALE, trans[n * 3 + 1] * POS_SCALE,
                    trans[n * 3 + 2] * POS_SCALE };
    float G[8][9], Gt[8][3];
    for (int g = 0; g < 8; ++g)
        for (int i = 0; i < 3; ++i) {
            for (int j = 0; j < 3; ++j)
                G[g][i * 3 + j] = Rn[i * 3 + 0] * Rl[g][0 + j] + Rn[i * 3 + 1] * Rl[g][3 + j]
                                + Rn[i * 3 + 2] * Rl[g][6 + j];
            Gt[g][i] = Rn[i * 3 + 0] * Tl[g][0] + Rn[i * 3 + 1] * Tl[g][1]
                     + Rn[i * 3 + 2] * Tl[g][2] + tw[i];
        }
    for (int a = 0; a < 14; ++a) {
        int g = a14g[aa * 14 + a];
        const float* lit = a14p + (int64_t)(aa * 14 + a) * 3;
        float mv = a14m[aa * 14 + a];
        for (int i = 0; i < 3; ++i) {
            float p = G[g][i * 3 + 0] * lit[0] + G[g][i * 3 + 1] * lit[1]
                    + G[g][i * 3 + 2] * lit[2] + Gt[g][i];
            pos_out[n * 42 + a * 3 + i] = p * mv;
        }
    }
}

// ---------------------------------------------------------------------------
// host side
// ---------------------------------------------------------------------------
static void gemm_f32(hipStream_t s, const float* A, const float* B, const float* bias,
                     float* C, int M, int N, int K, int lda, int ldb, int ldc,
                     int64_t sA, int64_t sB, int64_t sC, int batch, int transB, int relu)
{
    dim3 g(CDIV(M, 16), CDIV(N, 64), batch);
    k_gemm_f32<<<g, dim3(128), 0, s>>>(A, B, bias, C, M, N, K, lda, ldb, ldc,
                                       sA, sB, sC, transB, relu);
}

static void gemm_bf16(hipStream_t s, const __bf16* A, const __bf16* B, const float* bias,
                      float* C, int M, int N, int K, int lda, int ldb, int ldc,
                      int64_t sA, int64_t sB, int64_t sC, int batch)
{
    dim3 g(CDIV(M, 16), CDIV(N, 64), batch);
    k_gemm_bf16<<<g, dim3(128), 0, s>>>(A, B, bias, C, M, N, K, lda, ldb, ldc, sA, sB, sC);
}

extern "C" void kernel_launch(void* const* d_in, const int* in_sizes, int n_in,
                              void* d_out, int out_size, void* d_ws, size_t ws_size,
                              hipStream_t stream)
{
    (void)in_sizes; (void)n_in; (void)out_size; (void)ws_size;
    const float* single = (const float*)d_in[0];
    const float* pair   = (const float*)d_in[1];
    const float* mask   = (const float*)d_in[2];
    const int*   aatype = (const int*)d_in[3];
#define PF(i) ((const float*)d_in[i])
    const int* a14g = (const int*)d_in[54];
    float* out = (float*)d_out;
    float* trajOut = out + 196608;           // 8*512*7
    float* posOut  = out + 225280;           // 512*14*3
    float* angOut  = out + 246784;           // 512*7*2

    uint8_t* wsb = (uint8_t*)d_ws;
    size_t off = 0;
    auto alloc = [&](size_t bytes) -> void* {
        void* p = wsb + off;
        off += (bytes + 255) & ~(size_t)255;
        return p;
    };
    float*  initAct = (float*)alloc(196608 * 4);
    float*  act     = (float*)alloc(196608 * 4);
    float*  delta   = (float*)alloc(196608 * 4);
    float*  h0      = (float*)alloc(196608 * 4);
    float*  h1      = (float*)alloc(196608 * 4);
    __bf16* pairb   = (__bf16*)alloc((size_t)NRES * NRES * 128 * 2);
    __bf16* pairWb  = (__bf16*)alloc(1536 * 2);
    float*  b2d     = (float*)alloc((size_t)NRES * NRES * HH * 4);
    float*  qsB     = (float*)alloc(98304 * 4);
    float*  ksB     = (float*)alloc(98304 * 4);
    float*  vsB     = (float*)alloc(98304 * 4);
    float*  qp      = (float*)alloc(73728 * 4);
    float*  kp      = (float*)alloc(73728 * 4);
    float*  vp      = (float*)alloc(147456 * 4);
    float*  qq      = (float*)alloc(6144 * 4);
    float*  kk      = (float*)alloc(6144 * 4);
    float*  logS    = (float*)alloc((size_t)HH * NRES * NRES * 4);
    float*  qkP     = (float*)alloc((size_t)HH * NRES * NRES * 4);
    float*  attn    = (float*)alloc((size_t)NRES * HH * NRES * 4);
    __bf16* attnb   = (__bf16*)alloc((size_t)NRES * HH * NRES * 2);
    float*  resS    = (float*)alloc(98304 * 4);
    float*  resPg   = (float*)alloc(147456 * 4);
    float*  res2d   = (float*)alloc(786432 * 4);
    float*  feat    = (float*)alloc((size_t)NRES * 2112 * 4);
    float*  upd     = (float*)alloc(3072 * 4);
    float*  quat    = (float*)alloc(2048 * 4);
    float*  rotb    = (float*)alloc(4608 * 4);
    float*  transb  = (float*)alloc(1536 * 4);
    float*  scA     = (float*)alloc(65536 * 4);
    float*  scB     = (float*)alloc(65536 * 4);
    float*  scC     = (float*)alloc(65536 * 4);
    float*  scT     = (float*)alloc(65536 * 4);
    float*  un      = (float*)alloc(7168 * 4);

    // prologue: init_act = LN(single); act = init_act @ init_proj
    k_ln<<<NRES, 128, 0, stream>>>(single, nullptr, PF(4), PF(5), initAct, 384);
    gemm_f32(stream, initAct, PF(12), PF(13), act, NRES, 384, 384, 384, 384, 384, 0, 0, 0, 1, 0, 0);
    // pair_n (bf16), layer-invariant pair bias b2d = pair_n @ Wpair + b
    k_ln_pair_bf16<<<NRES * NRES, 128, 0, stream>>>(pair, PF(6), PF(7), pairb);
    k_cvt_bf16<<<CDIV(1536, 256), 256, 0, stream>>>(PF(27), pairWb, 1536);
    gemm_bf16(stream, pairb, pairWb, PF(28), b2d, NRES * NRES, 12, 128, 128, 12, 12, 0, 0, 0, 1);
    k_init_pose<<<CDIV(NRES, 128), 128, 0, stream>>>(quat, rotb, transb);

    for (int layer = 0; layer < 8; ++layer) {
        // projections
        gemm_f32(stream, act, PF(14), PF(15), qsB, NRES, 192, 384, 384, 192, 192, 0, 0, 0, 1, 0, 0);
        gemm_f32(stream, act, PF(16), PF(17), ksB, NRES, 192, 384, 384, 192, 192, 0, 0, 0, 1, 0, 0);
        gemm_f32(stream, act, PF(18), PF(19), vsB, NRES, 192, 384, 384, 192, 192, 0, 0, 0, 1, 0, 0);
        gemm_f32(stream, act, PF(20), PF(21), qp, NRES, 144, 384, 384, 144, 144, 0, 0, 0, 1, 0, 0);
        gemm_f32(stream, act, PF(22), PF(23), kp, NRES, 144, 384, 384, 144, 144, 0, 0, 0, 1, 0, 0);
        gemm_f32(stream, act, PF(24), PF(25), vp, NRES, 288, 384, 384, 288, 288, 0, 0, 0, 1, 0, 0);
        k_pts_global<<<CDIV(NRES * 48, 256), 256, 0, stream>>>(qp, rotb, transb, 48);
        k_pts_global<<<CDIV(NRES * 48, 256), 256, 0, stream>>>(kp, rotb, transb, 48);
        k_pts_global<<<CDIV(NRES * 96, 256), 256, 0, stream>>>(vp, rotb, transb, 96);
        k_sqsum12<<<CDIV(6144, 256), 256, 0, stream>>>(qp, qq, 6144);
        k_sqsum12<<<CDIV(6144, 256), 256, 0, stream>>>(kp, kk, 6144);
        // logits: scalar qs.ks (batched over h, B transposed), point qk (K=12->pad)
        gemm_f32(stream, qsB, ksB, nullptr, logS, NRES, NRES, 16, 192, 192, NRES,
                 16, 16, (int64_t)NRES * NRES, HH, 1, 0);
        gemm_f32(stream, qp, kp, nullptr, qkP, NRES, NRES, 12, 144, 144, NRES,
                 12, 12, (int64_t)NRES * NRES, HH, 1, 0);
        k_softmax<<<NRES * HH, 32, 0, stream>>>(logS, qkP, qq, kk, b2d, PF(26), mask, attn, attnb);
        // weighted sums
        gemm_f32(stream, attn, vsB, nullptr, resS, NRES, 16, NRES, HH * NRES, 192, 16,
                 NRES, 16, (int64_t)NRES * 16, HH, 0, 0);
        gemm_f32(stream, attn, vp, nullptr, resPg, NRES, 24, NRES, HH * NRES, 288, 24,
                 NRES, 24, (int64_t)NRES * 24, HH, 0, 0);
        k_res2d_async<<<dim3(2, NRES), 128, 0, stream>>>(attnb, pairb, res2d);
        k_feat<<<NRES, 256, 0, stream>>>(resS, resPg, res2d, rotb, transb, feat);
        // ipa out + residual LN
        gemm_f32(stream, feat, PF(29), PF(30), delta, NRES, 384, 2112, 2112, 384, 384, 0, 0, 0, 1, 0, 0);
        k_ln<<<NRES, 128, 0, stream>>>(act, delta, PF(8), PF(9), act, 384);
        // transition + residual LN
        gemm_f32(stream, act, PF(31), PF(32), h0, NRES, 384, 384, 384, 384, 384, 0, 0, 0, 1, 0, 1);
        gemm_f32(stream, h0, PF(33), PF(34), h1, NRES, 384, 384, 384, 384, 384, 0, 0, 0, 1, 0, 1);
        gemm_f32(stream, h1, PF(35), PF(36), delta, NRES, 384, 384, 384, 384, 384, 0, 0, 0, 1, 0, 0);
        k_ln<<<NRES, 128, 0, stream>>>(act, delta, PF(10), PF(11), act, 384);
        // affine update + compose + traj
        k_small_dense<<<CDIV(NRES * 6, 256), 256, 0, stream>>>(act, PF(37), PF(38), upd, 384, 6);
        k_compose<<<CDIV(NRES, 128), 128, 0, stream>>>(quat, rotb, transb, upd,
                                                       trajOut + (int64_t)layer * NRES * 7);
    }

    // sidechain (only final-layer results are observable)
    gemm_f32(stream, act, PF(39), PF(40), scA, NRES, 128, 384, 384, 128, 128, 0, 0, 0, 1, 0, 0);
    gemm_f32(stream, initAct, PF(41), PF(42), scB, NRES, 128, 384, 384, 128, 128, 0, 0, 0, 1, 0, 0);
    k_add<<<CDIV(65536, 256), 256, 0, stream>>>(scA, scB, 65536);
    for (int blk = 0; blk < 2; ++blk) {
        const float* w1 = PF(43 + blk * 4), *b1 = PF(44 + blk * 4);
        const float* w2 = PF(45 + blk * 4), *b2 = PF(46 + blk * 4);
        k_relu_copy<<<CDIV(65536, 256), 256, 0, stream>>>(scT, scA, 65536);
        gemm_f32(stream, scT, w1, b1, scB, NRES, 128, 128, 128, 128, 128, 0, 0, 0, 1, 0, 1);
        gemm_f32(stream, scB, w2, b2, scC, NRES, 128, 128, 128, 128, 128, 0, 0, 0, 1, 0, 0);
        k_add<<<CDIV(65536, 256), 256, 0, stream>>>(scA, scC, 65536);
    }
    k_relu_copy<<<CDIV(65536, 256), 256, 0, stream>>>(scT, scA, 65536);
    k_small_dense<<<CDIV(NRES * 14, 256), 256, 0, stream>>>(scT, PF(51), PF(52), un, 128, 14);
    k_sidechain<<<CDIV(NRES, 64), 64, 0, stream>>>(un, aatype, PF(53), a14g, PF(55), PF(56),
                                                   rotb, transb, angOut, posOut);
    // final act -> output
    hipMemcpyAsync(out, act, (size_t)196608 * 4, hipMemcpyDeviceToDevice, stream);
#undef PF
}
